// MACE_89979564851571
// MI455X (gfx1250) — compile-verified, matrix operand order checked by reading, not run on hardware
//
#include <hip/hip_runtime.h>
#include <cmath>

// ---------------------------------------------------------------------------
// MACE-like message passing for MI455X (gfx1250, wave32, WMMA + TDM).
//  k0: zero A accumulator + graph output
//  k1: node embedding  h = attrs@W_embed, e0 = attrs@atomic_energies
//  k2: edge kernel: geometry -> radial MLP (V_WMMA_F32_16X16X4_F32 tiles,
//      rW1/rW2/rW3 staged to LDS via TENSOR_LOAD_TO_LDS) -> fused msg
//      scatter (global_atomic_add_f32 into A, resident in the 192MB L2)
//  k3: node kernel: 9x [16,32]@[32,32] WMMA tiles (W_mix staged via TDM)
//      -> quadratic + readout -> atomic add into graph energies
// msg ([E,32,9] = 921 MB) is never materialized; A (57.6 MB) stays in L2.
// ---------------------------------------------------------------------------

#define WAVES 8
#define TPB 256

typedef __attribute__((ext_vector_type(8))) float v8f;

#if __has_builtin(__builtin_amdgcn_wmma_f32_16x16x4_f32)
// Native fp32 WMMA path: matches reference precision.
typedef __attribute__((ext_vector_type(2))) float frag_t;
constexpr int K_STEP = 4;
// A 16x4 f32 layout (ISA 7.12.2): lanes 0-15 = rows, VGPR0/1 = K {0,1};
// lanes 16-31 same rows, K {2,3}.  B mirrored (K rows across hi).
template <class L>
__device__ __forceinline__ frag_t make_frag(L ld, int kb, int hi) {
  frag_t f;
  f.x = ld(kb + 2 * hi + 0);
  f.y = ld(kb + 2 * hi + 1);
  return f;
}
__device__ __forceinline__ v8f wmma_step(frag_t a, frag_t b, v8f c) {
  return __builtin_amdgcn_wmma_f32_16x16x4_f32(false, a, false, b, (short)0, c,
                                               false, false);
}
#else
// Fallback: f16 inputs, f32 accumulate (codegen-confirmed builtin).
typedef __attribute__((ext_vector_type(16))) _Float16 frag_t;
constexpr int K_STEP = 32;
__device__ __forceinline__ int kmap(int i, int hi) {
  return (i < 8) ? (8 * hi + i) : (8 * hi + i + 8);
}
template <class L>
__device__ __forceinline__ frag_t make_frag(L ld, int kb, int hi) {
  frag_t f;
#pragma unroll
  for (int i = 0; i < 16; ++i) f[i] = (_Float16)ld(kb + kmap(i, hi));
  return f;
}
__device__ __forceinline__ v8f wmma_step(frag_t a, frag_t b, v8f c) {
  return __builtin_amdgcn_wmma_f32_16x16x32_f16(false, a, false, b, (short)0, c,
                                                false, false);
}
#endif

// Fast silu: v_rcp_f32 instead of the IEEE div expansion (activation gate,
// 1-ulp rcp is plenty).
__device__ __forceinline__ float silu(float v) {
  return v * __builtin_amdgcn_rcpf(1.f + __expf(-v));
}

// ---------------------------------------------------------------------------
// Tensor Data Mover: stage a contiguous f32 array (global -> LDS).
// Descriptor per CDNA5 ISA ch8: group0 = {count|flags, lds_addr, gaddr_lo,
// gaddr_hi|type=2}; group1 carries data_size/tile/tensor dims (bit-packed).
#if __has_builtin(__builtin_amdgcn_tensor_load_to_lds) && \
    __has_builtin(__builtin_amdgcn_s_wait_tensorcnt)
#define HAVE_TDM 1
typedef unsigned int u32x4 __attribute__((ext_vector_type(4)));
typedef int i32x4 __attribute__((ext_vector_type(4)));
typedef int i32x8 __attribute__((ext_vector_type(8)));

// Copy nelem f32 (nelem multiple of 256) as a 256 x (nelem/256) row-major tile.
__device__ __forceinline__ void tdm_load_f32(void* lds_dst, const void* gsrc,
                                             unsigned nelem) {
  const unsigned long long ga = (unsigned long long)(uintptr_t)gsrc;
  const unsigned lo = (unsigned)(uintptr_t)lds_dst;  // LDS byte offset
  const unsigned W = 256u, H = nelem / 256u;
  u32x4 g0;
  g0[0] = 1u;                                     // count=1 (valid user D#)
  g0[1] = lo;                                     // lds_addr
  g0[2] = (unsigned)ga;                           // global_addr[31:0]
  g0[3] = (unsigned)(ga >> 32) | (2u << 30);      // global_addr[56:32]|type=2
  i32x8 g1;
  g1[0] = (int)(2u << 16);                        // data_size=2 (4 bytes)
  g1[1] = (int)((W & 0xffffu) << 16);             // tensor_dim0[15:0] @bit48
  g1[2] = (int)((W >> 16) | ((H & 0xffffu) << 16));  // td0[31:16], td1[15:0]
  g1[3] = (int)((H >> 16) | (W << 16));           // td1[31:16] | tile_dim0
  g1[4] = (int)(H & 0xffffu);                     // tile_dim1 (tile_dim2=0)
  g1[5] = (int)W;                                 // tensor_dim0_stride lo32
  g1[6] = 0;                                      // stride0 hi | stride1 lo
  g1[7] = 0;
  i32x4 z4 = {0, 0, 0, 0};
#if defined(__clang_major__) && __clang_major__ >= 23
  i32x8 z8 = {0, 0, 0, 0, 0, 0, 0, 0};
  __builtin_amdgcn_tensor_load_to_lds(g0, g1, z4, z4, z8, 0);
#else
  __builtin_amdgcn_tensor_load_to_lds(g0, g1, z4, z4, 0);
#endif
}
#else
#define HAVE_TDM 0
#endif

// ---------------------------------------------------------------------------
__global__ void zero_f32(float* __restrict__ p, long n) {
  long i = (long)blockIdx.x * blockDim.x + threadIdx.x;
  long stride = (long)gridDim.x * blockDim.x;
  for (; i < n; i += stride) p[i] = 0.f;
}

// ---------------------------------------------------------------------------
__global__ void embed_kernel(const float* __restrict__ attrs,
                             const float* __restrict__ W_embed,
                             const float* __restrict__ ae,
                             float* __restrict__ h, float* __restrict__ e0,
                             int N, int S) {
  int n = blockIdx.x * blockDim.x + threadIdx.x;
  if (n >= N) return;
  float acc[32];
#pragma unroll
  for (int c = 0; c < 32; ++c) acc[c] = 0.f;
  float e = 0.f;
  for (int s = 0; s < S; ++s) {
    float a = attrs[(long)n * S + s];
    e += a * ae[s];
#pragma unroll
    for (int c = 0; c < 32; ++c) acc[c] += a * W_embed[s * 32 + c];
  }
#pragma unroll
  for (int c = 0; c < 32; ++c) h[(long)n * 32 + c] = acc[c];
  e0[n] = e;
}

// ---------------------------------------------------------------------------
// Edge kernel: each wave owns a 16-edge tile.  All radial-MLP weights are
// staged once per block into LDS by the Tensor Data Mover; inner-loop
// B-fragments then come from conflict-free ds_load instead of global.
__global__ __launch_bounds__(TPB) void mace_edge_kernel(
    const float* __restrict__ pos, const float* __restrict__ shifts,
    const int* __restrict__ eidx, const float* __restrict__ h,
    const float* __restrict__ rW1, const float* __restrict__ rb1,
    const float* __restrict__ rW2, const float* __restrict__ rb2,
    const float* __restrict__ rW3, float* __restrict__ Aacc, int E) {
  __shared__ float s_w1[8 * 64];          // 2 KB
  __shared__ float s_w2[64 * 64];         // 16 KB
  __shared__ float s_w3[64 * 96];         // 24 KB
  __shared__ float s_h[WAVES][16 * 64];   // h1/h2 staging (C-layout->A-layout)
  __shared__ float s_hs[WAVES][16 * 32];  // h[sender] per edge row
  __shared__ float s_Y[WAVES][16 * 9];    // spherical harmonics per edge row
  __shared__ int s_recv[WAVES][16];       // receiver (or -1)

  const int wave = threadIdx.x >> 5;
  const int lane = threadIdx.x & 31;
  const int m = lane & 15;   // tile row == edge within tile
  const int hi = lane >> 4;  // K-half selector for A/B fragments
  const long tile = (long)blockIdx.x * WAVES + wave;
  const long e = tile * 16 + m;
  const bool valid = e < (long)E;

  // Stage weights into LDS (one DMA wave; TDM ops are in-order per wave,
  // a single s_wait_tensorcnt 0 covers all three descriptors).
#if HAVE_TDM
  if (threadIdx.x < 32) {
    tdm_load_f32(s_w1, rW1, 512u);
    tdm_load_f32(s_w2, rW2, 4096u);
    tdm_load_f32(s_w3, rW3, 6144u);
    __builtin_amdgcn_s_wait_tensorcnt(0);
  }
#else
  for (int i = threadIdx.x; i < 512; i += TPB) s_w1[i] = rW1[i];
  for (int i = threadIdx.x; i < 4096; i += TPB) s_w2[i] = rW2[i];
  for (int i = threadIdx.x; i < 6144; i += TPB) s_w3[i] = rW3[i];
#endif

  // ---- geometry (lanes 16-31 duplicate their row's edge) ----
  float px = 0.f, py = 0.f, pz = 0.f;
  int snd = 0, rcv = -1;
  if (valid) {
    snd = eidx[e];
    rcv = eidx[(long)E + e];
    px = pos[3 * rcv + 0] - pos[3 * snd + 0] + shifts[3 * e + 0];
    py = pos[3 * rcv + 1] - pos[3 * snd + 1] + shifts[3 * e + 1];
    pz = pos[3 * rcv + 2] - pos[3 * snd + 2] + shifts[3 * e + 2];
  }
  const float len = sqrtf(px * px + py * py + pz * pz);
  const float safe_r = fmaxf(len, 1e-9f);
  const float inv = 1.f / safe_r;
  const float ux = px * inv, uy = py * inv, uz = pz * inv;
  // polynomial cutoff, p = 6
  const float u = len * 0.2f;  // r / R_MAX
  const float u2 = u * u, u6 = u2 * u2 * u2, u7 = u6 * u, u8 = u7 * u;
  float fcut = 1.f - 28.f * u6 + 48.f * u7 - 21.f * u8;
  fcut = (u < 1.f) ? fcut : 0.f;
  // bessel prefactor: sqrt(2/R_MAX)/safe_r * fcut; rb[k] = pref*sin((k+1)*warg)
  const float pref = valid ? (0.632455532f * inv * fcut) : 0.f;  // sqrt(0.4)
  const float warg = 3.14159265358979f * safe_r * 0.2f;

  if (hi == 0) {
    s_recv[wave][m] = valid ? rcv : -1;
    const float s3 = 1.73205080757f, s5 = 2.2360679775f, s15 = 3.87298334621f;
    float* Yp = &s_Y[wave][m * 9];
    Yp[0] = 1.f;
    Yp[1] = s3 * ux;
    Yp[2] = s3 * uy;
    Yp[3] = s3 * uz;
    Yp[4] = s15 * ux * uy;
    Yp[5] = s15 * uy * uz;
    Yp[6] = 0.5f * s5 * (3.f * uz * uz - 1.f);
    Yp[7] = s15 * ux * uz;
    Yp[8] = 0.5f * s15 * (ux * ux - uy * uy);
    if (valid) {
      const float* hp = &h[(long)snd * 32];
#pragma unroll
      for (int c = 0; c < 32; ++c) s_hs[wave][m * 32 + c] = hp[c];
    }
  }
  __syncthreads();  // publishes TDM-staged weights + per-wave staging

  // ---- layer 1: h1 = silu(rb[16x8] @ rW1[8x64] + rb1) ----
  {
    v8f acc[4] = {};
#pragma unroll
    for (int kb = 0; kb < 8; kb += K_STEP) {
      frag_t af = make_frag(
          [&](int k) {
            return (k < 8) ? pref * __sinf((float)(k + 1) * warg) : 0.f;
          },
          kb, hi);
#pragma unroll
      for (int t = 0; t < 4; ++t) {
        const int col = t * 16 + m;
        frag_t bf = make_frag(
            [&](int k) { return (k < 8) ? s_w1[k * 64 + col] : 0.f; }, kb, hi);
        acc[t] = wmma_step(af, bf, acc[t]);
      }
    }
#pragma unroll
    for (int t = 0; t < 4; ++t) {
      const int col = t * 16 + m;
      const float bias = rb1[col];
#pragma unroll
      for (int j = 0; j < 8; ++j)
        s_h[wave][(j + 8 * hi) * 64 + col] = silu(acc[t][j] + bias);
    }
  }
  __syncthreads();

  // ---- layer 2: h2 = silu(h1[16x64] @ rW2[64x64] + rb2) ----
  {
    v8f acc[4] = {};
    for (int kb = 0; kb < 64; kb += K_STEP) {
      frag_t af =
          make_frag([&](int k) { return s_h[wave][m * 64 + k]; }, kb, hi);
#pragma unroll
      for (int t = 0; t < 4; ++t) {
        const int col = t * 16 + m;
        frag_t bf =
            make_frag([&](int k) { return s_w2[k * 64 + col]; }, kb, hi);
        acc[t] = wmma_step(af, bf, acc[t]);
      }
    }
    __syncthreads();  // all h1 reads done before overwrite
#pragma unroll
    for (int t = 0; t < 4; ++t) {
      const int col = t * 16 + m;
      const float bias = rb2[col];
#pragma unroll
      for (int j = 0; j < 8; ++j)
        s_h[wave][(j + 8 * hi) * 64 + col] = silu(acc[t][j] + bias);
    }
  }
  __syncthreads();

  // ---- layer 3: R = h2[16x64] @ rW3[64x96]; fused msg scatter ----
  {
    v8f acc[6] = {};
    for (int kb = 0; kb < 64; kb += K_STEP) {
      frag_t af =
          make_frag([&](int k) { return s_h[wave][m * 64 + k]; }, kb, hi);
#pragma unroll
      for (int t = 0; t < 6; ++t) {
        const int col = t * 16 + m;
        frag_t bf =
            make_frag([&](int k) { return s_w3[k * 96 + col]; }, kb, hi);
        acc[t] = wmma_step(af, bf, acc[t]);
      }
    }
    // C layout: VGPR j holds row (j + 8*hi), column = t*16 + m.
    // Column ncol of [E,96] maps to (c = ncol/3, l = ncol%3);
    // l -> lm group: {0}, {1,2,3}, {4..8}.
#pragma unroll
    for (int t = 0; t < 6; ++t) {
      const int ncol = t * 16 + m;
      const int c = ncol / 3;
      const int l = ncol - 3 * c;
      const int lm0 = (l == 0) ? 0 : ((l == 1) ? 1 : 4);
      const int lcnt = (l == 0) ? 1 : ((l == 1) ? 3 : 5);
#pragma unroll
      for (int j = 0; j < 8; ++j) {
        const int row = j + 8 * hi;
        const int rr = s_recv[wave][row];
        if (rr >= 0) {
          const float base = acc[t][j] * s_hs[wave][row * 32 + c];
          float* dst = &Aacc[(long)rr * 288 + c * 9 + lm0];
          const float* Yp = &s_Y[wave][row * 9 + lm0];
          for (int q = 0; q < lcnt; ++q) atomicAdd(dst + q, base * Yp[q]);
        }
      }
    }
  }
}

// ---------------------------------------------------------------------------
// Node kernel: each wave owns a 16-node tile; 9 per-lm [16,32]@[32,32] GEMMs.
// W_mix (12 KB) is TDM-staged into LDS once per block.
__global__ __launch_bounds__(TPB) void mace_node_kernel(
    const float* __restrict__ Aacc, const float* __restrict__ W_mix,
    const float* __restrict__ w_quad, const float* __restrict__ W_readout,
    const float* __restrict__ e0, const int* __restrict__ batch,
    float* __restrict__ graph_e, int N) {
  __shared__ float s_wm[3 * 32 * 32];  // 12 KB
  __shared__ float s_b[WAVES][16 * 32];
  const int wave = threadIdx.x >> 5;
  const int lane = threadIdx.x & 31;
  const int m = lane & 15;
  const int hi = lane >> 4;
  const long tile = (long)blockIdx.x * WAVES + wave;
  const int n = (int)(tile * 16) + m;
  const bool valid = tile * 16 + m < (long)N;

#if HAVE_TDM
  if (threadIdx.x < 32) {
    tdm_load_f32(s_wm, W_mix, 3072u);
    __builtin_amdgcn_s_wait_tensorcnt(0);
  }
#else
  for (int i = threadIdx.x; i < 3072; i += TPB) s_wm[i] = W_mix[i];
#endif
  __syncthreads();

  v8f bacc[2] = {};  // B[n,d] accumulator in C layout
  for (int lm = 0; lm < 9; ++lm) {
    const int l = (lm == 0) ? 0 : ((lm < 4) ? 1 : 2);
    const float wq = w_quad[l];
    v8f acc[2] = {};
    for (int kb = 0; kb < 32; kb += K_STEP) {
      // A-fragment: A[n, c=k, lm] / AVG_NEIGH
      frag_t af = make_frag(
          [&](int k) {
            return valid ? Aacc[((long)n * 32 + k) * 9 + lm] * 0.0625f : 0.f;
          },
          kb, hi);
#pragma unroll
      for (int t = 0; t < 2; ++t) {
        const int col = t * 16 + m;
        frag_t bf = make_frag(
            [&](int k) { return s_wm[(l * 32 + k) * 32 + col]; }, kb, hi);
        acc[t] = wmma_step(af, bf, acc[t]);
      }
    }
    // B += (lm==0 ? A2 : 0) + w_quad[l] * A2^2
#pragma unroll
    for (int t = 0; t < 2; ++t)
#pragma unroll
      for (int j = 0; j < 8; ++j) {
        const float a2 = acc[t][j];
        float add = wq * a2 * a2;
        if (lm == 0) add += a2;
        bacc[t][j] += add;
      }
  }
  // readout: node_e = sum_d B[n,d]*W_readout[d] + e0[n]; reduce via LDS.
#pragma unroll
  for (int t = 0; t < 2; ++t) {
    const int col = t * 16 + m;
    const float wr = W_readout[col];
#pragma unroll
    for (int j = 0; j < 8; ++j)
      s_b[wave][(j + 8 * hi) * 32 + col] = bacc[t][j] * wr;
  }
  __syncthreads();
  if (hi == 0 && valid) {
    float s = 0.f;
#pragma unroll
    for (int d = 0; d < 32; ++d) s += s_b[wave][m * 32 + d];
    s += e0[n];
    atomicAdd(&graph_e[batch[n]], s);
  }
}

// ---------------------------------------------------------------------------
extern "C" void kernel_launch(void* const* d_in, const int* in_sizes, int n_in,
                              void* d_out, int out_size, void* d_ws,
                              size_t ws_size, hipStream_t stream) {
  const float* node_attrs = (const float*)d_in[0];
  const float* positions = (const float*)d_in[1];
  const float* shifts = (const float*)d_in[2];
  const float* W_embed = (const float*)d_in[3];
  const float* rW1 = (const float*)d_in[4];
  const float* rb1 = (const float*)d_in[5];
  const float* rW2 = (const float*)d_in[6];
  const float* rb2 = (const float*)d_in[7];
  const float* rW3 = (const float*)d_in[8];
  const float* W_mix = (const float*)d_in[9];
  const float* w_quad = (const float*)d_in[10];
  const float* W_readout = (const float*)d_in[11];
  const float* atomic_energies = (const float*)d_in[12];
  const int* edge_index = (const int*)d_in[13];
  const int* batch = (const int*)d_in[14];
  float* graph_e = (float*)d_out;

  const int N = in_sizes[1] / 3;   // nodes
  const int E = in_sizes[13] / 2;  // edges
  const int S = in_sizes[12];      // species

  // workspace: h[N*32] | e0[N] | A[N*288]  (~64 MB total)
  float* h = (float*)d_ws;
  float* e0 = h + (long)N * 32;
  float* Aacc = e0 + N;

  zero_f32<<<2048, TPB, 0, stream>>>(Aacc, (long)N * 288);
  zero_f32<<<1, TPB, 0, stream>>>(graph_e, (long)out_size);

  embed_kernel<<<(N + TPB - 1) / TPB, TPB, 0, stream>>>(
      node_attrs, W_embed, atomic_energies, h, e0, N, S);

  const int etiles = (E + 15) / 16;
  mace_edge_kernel<<<(etiles + WAVES - 1) / WAVES, TPB, 0, stream>>>(
      positions, shifts, edge_index, h, rW1, rb1, rW2, rb2, rW3, Aacc, E);

  const int ntiles = (N + 15) / 16;
  mace_node_kernel<<<(ntiles + WAVES - 1) / WAVES, TPB, 0, stream>>>(
      Aacc, W_mix, w_quad, W_readout, e0, batch, graph_e, N);
}
